// RotaryAttention_41807211659771
// MI455X (gfx1250) — matrix operand order repaired and checked by reading, not compiled, not run on hardware
//
#include <hip/hip_runtime.h>
#include <math.h>

// ---------------- CDNA5 WMMA types ----------------
typedef __attribute__((ext_vector_type(16))) __bf16 v16bf;
typedef __attribute__((ext_vector_type(8)))  float  v8f;
typedef __attribute__((ext_vector_type(4)))  unsigned int u32x4;
typedef __attribute__((ext_vector_type(8)))  int i32x8;
typedef __attribute__((ext_vector_type(4)))  int i32x4;

#define D_MODEL 1024
#define NHEAD   16
#define DK      64
#define TSEQ    2048
#define BATCH   4
#define MTOT    (BATCH * TSEQ)   // 8192

#if defined(__has_builtin)
#if __has_builtin(__builtin_amdgcn_tensor_load_to_lds)
#define HAVE_TDM 1
#endif
#if __has_builtin(__builtin_amdgcn_cvt_pk_bf16_f32)
#define HAVE_PK_BF16 1
#endif
#endif

static __device__ __forceinline__ unsigned short f2bf(float f) {
  unsigned int u = __builtin_bit_cast(unsigned int, f);
  u += 0x7FFFu + ((u >> 16) & 1u);           // round-to-nearest-even
  return (unsigned short)(u >> 16);
}
static __device__ __forceinline__ __bf16 bfc(unsigned short s) {
  return __builtin_bit_cast(__bf16, s);
}
#ifdef HAVE_PK_BF16
typedef __attribute__((ext_vector_type(2))) __bf16 bf16x2;
#endif
// pack two f32 -> two bf16 in one uint (v_cvt_pk_bf16_f32 when available)
static __device__ __forceinline__ unsigned int f2bf2(float lo, float hi) {
#ifdef HAVE_PK_BF16
  bf16x2 p = __builtin_amdgcn_cvt_pk_bf16_f32(lo, hi);
  return __builtin_bit_cast(unsigned int, p);
#else
  return (unsigned)f2bf(lo) | ((unsigned)f2bf(hi) << 16);
#endif
}

static __device__ __forceinline__ void wait_tensorcnt0() {
#if defined(__has_builtin)
#if __has_builtin(__builtin_amdgcn_s_wait_tensorcnt)
  __builtin_amdgcn_s_wait_tensorcnt(0);
#else
  asm volatile("s_wait_tensorcnt 0x0" ::: "memory");
#endif
#endif
}

#ifdef HAVE_TDM
// TDM 2D tile load: bf16 (data_size=1 code), rows x cols elements, global row
// stride `row_stride` elements, LDS rows padded by pad_amount DWORDs every
// pad_interval-coded DWORDs (ISA 8.4: interval code k -> 2^(k+1) DWORDs,
// amount code k -> k+1 DWORDs).
static __device__ __forceinline__ void tdm_load_2d_bf16(
    const unsigned short* gptr, unsigned lds_byte_off,
    unsigned rows, unsigned cols, unsigned row_stride,
    unsigned pad_interval_code, unsigned pad_amount_code) {
  unsigned long long ga = (unsigned long long)(size_t)gptr;
  u32x4 g0;
  g0[0] = 1u;                                             // count=1
  g0[1] = lds_byte_off;                                   // lds_addr
  g0[2] = (unsigned)(ga & 0xFFFFFFFFu);                   // global_addr[31:0]
  g0[3] = (unsigned)((ga >> 32) & 0x1FFFFFFu) | (2u << 30); // ga[56:32], type=2
  const unsigned long long s0 = row_stride;               // tensor_dim0_stride
  i32x8 g1;
  g1[0] = (int)((1u << 16)                                // data_size = 2 bytes
              | (1u << 20)                                // pad_enable
              | (pad_interval_code << 22)
              | (pad_amount_code << 25));
  g1[1] = (int)((cols & 0xFFFFu) << 16);                  // tensor_dim0[15:0]
  g1[2] = (int)(((cols >> 16) & 0xFFFFu) | ((rows & 0xFFFFu) << 16)); // td0 hi, td1 lo
  g1[3] = (int)(((rows >> 16) & 0xFFFFu) | ((cols & 0xFFFFu) << 16)); // td1 hi, tile_dim0
  g1[4] = (int)(rows & 0xFFFFu);                          // tile_dim1 (tile_dim2=0)
  g1[5] = (int)(s0 & 0xFFFFFFFFu);                        // stride0[31:0]
  g1[6] = (int)((s0 >> 32) & 0xFFFFu);                    // stride0[47:32]
  g1[7] = 0;
  i32x4 z4 = {0, 0, 0, 0};
  i32x8 z8 = {0, 0, 0, 0, 0, 0, 0, 0};
  // 6-arg form (clang-23 / therock headers): g0, g1, g2, g3, extra group, cpol
  __builtin_amdgcn_tensor_load_to_lds(g0, g1, z4, z4, z8, 0);
}
#endif

// A fragment (16x32 bf16, row-major tile in LDS).
// ISA 7.12.2: lanes 0-15 -> M=lane, K in {0..7, 16..23}; lanes 16-31 -> M=lane-16, K in {8..15, 24..31}.
static __device__ __forceinline__ v16bf ldsA(const unsigned short* t, int stride) {
  int l  = threadIdx.x & 31;
  int r  = l & 15;
  int kb = (l >> 4) * 8;
  const unsigned short* p = t + r * stride + kb;
  v16bf f;
#pragma unroll
  for (int i = 0; i < 8; ++i) { f[i] = bfc(p[i]); f[i + 8] = bfc(p[i + 16]); }
  return f;
}

// B fragment (32x16 bf16). LDS tile is stored as B^T (i.e. [n][k] row-major).
// ISA: lanes 0-15 -> N=lane, K in {0..7, 16..23}; lanes 16-31 -> N=lane-16, K in {8..15, 24..31}.
static __device__ __forceinline__ v16bf ldsB(const unsigned short* t, int stride) {
  int l  = threadIdx.x & 31;
  int n  = l & 15;
  int kb = (l >> 4) * 8;
  const unsigned short* p = t + n * stride + kb;
  v16bf f;
#pragma unroll
  for (int i = 0; i < 8; ++i) { f[i] = bfc(p[i]); f[i + 8] = bfc(p[i + 16]); }
  return f;
}

static __device__ __forceinline__ v8f wmma_bf16(v16bf a, v16bf b, v8f c) {
  return __builtin_amdgcn_wmma_f32_16x16x32_bf16(false, a, false, b, (short)0, c, false, false);
}

// =====================================================================
// Kernel 1: C = X @ W^T + bias, optional fused RoPE, store bf16 [b][h][t][d]
// grid (MTOT/128, D_MODEL/64), block 256 (8 waves). N-tile 64 == one head.
// =====================================================================
__global__ __launch_bounds__(256)
void qkv_rope_kernel(const float* __restrict__ X, const float* __restrict__ W,
                     const float* __restrict__ bias,
                     const float* __restrict__ cosp, const float* __restrict__ sinp,
                     unsigned short* __restrict__ out, int do_rope) {
  __shared__ __align__(16) unsigned short As[128][40];
  __shared__ __align__(16) unsigned short Ws[64][40];

  const int m0  = blockIdx.x * 128;
  const int n0  = blockIdx.y * 64;
  const int tid = threadIdx.x;
  const int wv  = tid >> 5;

  v8f acc[4];
#pragma unroll
  for (int n = 0; n < 4; ++n)
#pragma unroll
    for (int j = 0; j < 8; ++j) acc[n][j] = 0.0f;

  for (int kc = 0; kc < D_MODEL / 32; ++kc) {
    const int k0 = kc * 32;
    // stage A tile 128x32 f32 -> bf16
#pragma unroll
    for (int it = 0; it < 4; ++it) {
      int idx = tid + 256 * it;              // 0..1023
      int r = idx >> 3, c4 = (idx & 7) * 4;
      const float* src = X + (size_t)(m0 + r) * D_MODEL + k0 + c4;
      float4 v = *(const float4*)src;
      *(unsigned int*)&As[r][c4 + 0] = f2bf2(v.x, v.y);
      *(unsigned int*)&As[r][c4 + 2] = f2bf2(v.z, v.w);
      if (kc + 1 < D_MODEL / 32) __builtin_prefetch(src + 32, 0, 1);
    }
    // stage W tile 64x32 (rows n0..n0+63) -> bf16 (this is B^T layout)
#pragma unroll
    for (int it = 0; it < 2; ++it) {
      int idx = tid + 256 * it;              // 0..511
      int r = idx >> 3, c4 = (idx & 7) * 4;
      const float* src = W + (size_t)(n0 + r) * D_MODEL + k0 + c4;
      float4 v = *(const float4*)src;
      *(unsigned int*)&Ws[r][c4 + 0] = f2bf2(v.x, v.y);
      *(unsigned int*)&Ws[r][c4 + 2] = f2bf2(v.z, v.w);
      if (kc + 1 < D_MODEL / 32) __builtin_prefetch(src + 32, 0, 1);
    }
    __syncthreads();
    v16bf af = ldsA(&As[wv * 16][0], 40);
#pragma unroll
    for (int n = 0; n < 4; ++n) {
      v16bf bf = ldsB(&Ws[n * 16][0], 40);
      acc[n] = wmma_bf16(af, bf, acc[n]);
    }
    __syncthreads();
  }

  // epilogue: bias, RoPE, store bf16 [b][h][t][d]
  const int l = tid & 31, half = l >> 4, nn = l & 15;
  const int head = n0 >> 6;
#pragma unroll
  for (int j = 0; j < 8; ++j) {
    const int m    = m0 + wv * 16 + (half ? j + 8 : j);
    const int b    = m >> 11;                 // /TSEQ
    const int tpos = m & (TSEQ - 1);
    float vals[4];
#pragma unroll
    for (int n = 0; n < 4; ++n) vals[n] = acc[n][j] + bias[n0 + n * 16 + nn];
    if (do_rope) {
      const float* cr = cosp + (size_t)tpos * DK;
      const float* sr = sinp + (size_t)tpos * DK;
      float rot[4];
#pragma unroll
      for (int n = 0; n < 4; ++n) {
        const int d = n * 16 + nn;
        const float partner = (n < 2) ? -vals[n + 2] : vals[n - 2];
        rot[n] = vals[n] * cr[d] + partner * sr[d];
      }
#pragma unroll
      for (int n = 0; n < 4; ++n) vals[n] = rot[n];
    }
    const size_t base = (((size_t)b * NHEAD + head) * TSEQ + tpos) * DK;
#pragma unroll
    for (int n = 0; n < 4; ++n) out[base + n * 16 + nn] = f2bf(vals[n]);
  }
}

// =====================================================================
// Kernel 2: flash attention per (b, h, 128-query tile). block 256 = 8 waves,
// each wave owns 16 query rows. Streams 64-key tiles. bf16 WMMA, f32 accum.
// Q/K tiles staged by the Tensor Data Mover when available.
// =====================================================================
__global__ __launch_bounds__(256)
void flash_attn_kernel(const unsigned short* __restrict__ Q,
                       const unsigned short* __restrict__ K,
                       const unsigned short* __restrict__ V,
                       const unsigned char* __restrict__ mask,
                       unsigned short* __restrict__ O) {
  __shared__ __align__(16) unsigned short Qs[128][72];
  __shared__ __align__(16) unsigned short Ks[64][72];
  __shared__ __align__(16) unsigned short VTs[64][72];   // [d][key]
  __shared__ __align__(16) unsigned short Ps[8][16][72]; // per-wave P scratch

  const int b  = blockIdx.z, h = blockIdx.y, q0 = blockIdx.x * 128;
  const int tid = threadIdx.x, wv = tid >> 5, l = tid & 31;
  const int half = l >> 4, nn = l & 15;
  const size_t headbase = (((size_t)b * NHEAD) + h) * (size_t)TSEQ * DK;

  // stage Q tile (128x64 bf16): TDM (pad 32-DWORD rows by 4 DWORDs -> stride 72)
#ifdef HAVE_TDM
  if (wv == 0)
    tdm_load_2d_bf16(Q + headbase + (size_t)q0 * DK,
                     (unsigned)(size_t)&Qs[0][0], 128, 64, 64, 4, 3);
#else
#pragma unroll
  for (int it = 0; it < 4; ++it) {
    int idx = tid + 256 * it;                // 0..1023
    int r = idx >> 3, s8 = (idx & 7) * 8;
    uint4 v = *(const uint4*)(Q + headbase + (size_t)(q0 + r) * DK + s8);
    *(uint4*)&Qs[r][s8] = v;
  }
#endif

  v8f acco[4];
  float mrow[8], lrow[8];
#pragma unroll
  for (int n = 0; n < 4; ++n)
#pragma unroll
    for (int j = 0; j < 8; ++j) acco[n][j] = 0.0f;
#pragma unroll
  for (int j = 0; j < 8; ++j) { mrow[j] = -__builtin_inff(); lrow[j] = 0.0f; }

  for (int kb = 0; kb < TSEQ / 64; ++kb) {
    const int k0 = kb * 64;
    // stage K tile (64x64)
#ifdef HAVE_TDM
    if (wv == 0)
      tdm_load_2d_bf16(K + headbase + (size_t)k0 * DK,
                       (unsigned)(size_t)&Ks[0][0], 64, 64, 64, 4, 3);
#else
#pragma unroll
    for (int it = 0; it < 2; ++it) {
      int idx = tid + 256 * it;              // 0..511
      int r = idx >> 3, s8 = (idx & 7) * 8;
      uint4 v = *(const uint4*)(K + headbase + (size_t)(k0 + r) * DK + s8);
      *(uint4*)&Ks[r][s8] = v;
    }
#endif
    // stage V tile transposed -> VTs[d][key] (manual: TDM cannot transpose)
#pragma unroll
    for (int it = 0; it < 8; ++it) {
      int idx = tid + 256 * it;              // 0..2047 uints
      int r = idx >> 5, d2 = idx & 31;
      unsigned int v = *(const unsigned int*)(V + headbase + (size_t)(k0 + r) * DK + d2 * 2);
      VTs[d2 * 2 + 0][r] = (unsigned short)(v & 0xFFFFu);
      VTs[d2 * 2 + 1][r] = (unsigned short)(v >> 16);
    }
#ifdef HAVE_TDM
    if (wv == 0) wait_tensorcnt0();          // Q (first iter) + K tile in LDS
#endif
    __syncthreads();

    // S = Q K^T for 16x64 strip
    v8f accs[4];
#pragma unroll
    for (int n = 0; n < 4; ++n)
#pragma unroll
      for (int j = 0; j < 8; ++j) accs[n][j] = 0.0f;
#pragma unroll
    for (int c = 0; c < 2; ++c) {
      v16bf af = ldsA(&Qs[wv * 16][c * 32], 72);
#pragma unroll
      for (int n = 0; n < 4; ++n) {
        v16bf bf = ldsB(&Ks[n * 16][c * 32], 72);
        accs[n] = wmma_bf16(af, bf, accs[n]);
      }
    }

    // scale + mask
    const float scale = 0.125f;               // 1/sqrt(64)
#pragma unroll
    for (int j = 0; j < 8; ++j) {
      const int qrow = q0 + wv * 16 + (half ? j + 8 : j);
#pragma unroll
      for (int n = 0; n < 4; ++n) {
        float s = accs[n][j] * scale;
        const int kcol = k0 + n * 16 + nn;
        if (mask[((size_t)b * TSEQ + qrow) * TSEQ + kcol]) s = -__builtin_inff();
        accs[n][j] = s;
      }
    }

    // online softmax (rows live in 16-lane halves; reduce with shfl_xor)
    float mnew[8], alpha[8];
#pragma unroll
    for (int j = 0; j < 8; ++j) {
      float mx = fmaxf(fmaxf(accs[0][j], accs[1][j]), fmaxf(accs[2][j], accs[3][j]));
      mx = fmaxf(mx, __shfl_xor(mx, 8, 32));
      mx = fmaxf(mx, __shfl_xor(mx, 4, 32));
      mx = fmaxf(mx, __shfl_xor(mx, 2, 32));
      mx = fmaxf(mx, __shfl_xor(mx, 1, 32));
      mnew[j]  = fmaxf(mrow[j], mx);
      alpha[j] = (mrow[j] == -__builtin_inff()) ? 0.0f : __expf(mrow[j] - mnew[j]);
    }
#pragma unroll
    for (int j = 0; j < 8; ++j) {
      const int prow = (half ? j + 8 : j);
      const bool dead = (mnew[j] == -__builtin_inff());
      float rs = 0.0f;
#pragma unroll
      for (int n = 0; n < 4; ++n) {
        float p = dead ? 0.0f : __expf(accs[n][j] - mnew[j]);
        rs += p;
        Ps[wv][prow][n * 16 + nn] = f2bf(p);
        acco[n][j] *= alpha[j];
      }
      rs += __shfl_xor(rs, 8, 32);
      rs += __shfl_xor(rs, 4, 32);
      rs += __shfl_xor(rs, 2, 32);
      rs += __shfl_xor(rs, 1, 32);
      lrow[j] = lrow[j] * alpha[j] + rs;
      mrow[j] = mnew[j];
    }

    // O += P @ V  (per-wave P from LDS, DS ops are in-order within a wave)
#pragma unroll
    for (int c = 0; c < 2; ++c) {
      v16bf af = ldsA(&Ps[wv][0][c * 32], 72);
#pragma unroll
      for (int n = 0; n < 4; ++n) {
        v16bf bf = ldsB(&VTs[n * 16][c * 32], 72);
        acco[n] = wmma_bf16(af, bf, acco[n]);
      }
    }
    __syncthreads();
  }

  // finalize: divide by l, store bf16 in [b][t][h*64+d] (GEMM-ready layout)
#pragma unroll
  for (int j = 0; j < 8; ++j) {
    const float inv = (lrow[j] > 0.0f) ? 1.0f / lrow[j] : 0.0f;
    const int t = q0 + wv * 16 + (half ? j + 8 : j);
    const size_t rb = (((size_t)b * TSEQ) + t) * D_MODEL + h * DK;
#pragma unroll
    for (int n = 0; n < 4; ++n) O[rb + n * 16 + nn] = f2bf(acco[n][j] * inv);
  }
}

// =====================================================================
// Kernel 3: out = A(bf16) @ Wo^T + bo  -> f32. A tile staged by TDM.
// =====================================================================
__global__ __launch_bounds__(256)
void oproj_kernel(const unsigned short* __restrict__ A, const float* __restrict__ W,
                  const float* __restrict__ bias, float* __restrict__ Cout) {
  __shared__ __align__(16) unsigned short As[128][40];
  __shared__ __align__(16) unsigned short Ws[64][40];

  const int m0 = blockIdx.x * 128, n0 = blockIdx.y * 64;
  const int tid = threadIdx.x, wv = tid >> 5;

  v8f acc[4];
#pragma unroll
  for (int n = 0; n < 4; ++n)
#pragma unroll
    for (int j = 0; j < 8; ++j) acc[n][j] = 0.0f;

  for (int kc = 0; kc < D_MODEL / 32; ++kc) {
    const int k0 = kc * 32;
    // stage A tile 128x32 bf16 (16-DWORD rows padded by 4 DWORDs -> stride 40)
#ifdef HAVE_TDM
    if (wv == 0)
      tdm_load_2d_bf16(A + (size_t)m0 * D_MODEL + k0,
                       (unsigned)(size_t)&As[0][0], 128, 32, D_MODEL, 3, 3);
#else
#pragma unroll
    for (int it = 0; it < 2; ++it) {
      int idx = tid + 256 * it;              // 0..511 uint4s
      int r = idx >> 2, s8 = (idx & 3) * 8;
      uint4 v = *(const uint4*)(A + (size_t)(m0 + r) * D_MODEL + k0 + s8);
      *(uint4*)&As[r][s8] = v;
    }
#endif
#pragma unroll
    for (int it = 0; it < 2; ++it) {
      int idx = tid + 256 * it;
      int r = idx >> 3, c4 = (idx & 7) * 4;
      const float* src = W + (size_t)(n0 + r) * D_MODEL + k0 + c4;
      float4 v = *(const float4*)src;
      *(unsigned int*)&Ws[r][c4 + 0] = f2bf2(v.x, v.y);
      *(unsigned int*)&Ws[r][c4 + 2] = f2bf2(v.z, v.w);
      if (kc + 1 < D_MODEL / 32) __builtin_prefetch(src + 32, 0, 1);
    }
#ifdef HAVE_TDM
    if (wv == 0) wait_tensorcnt0();
#endif
    __syncthreads();
    v16bf af = ldsA(&As[wv * 16][0], 40);
#pragma unroll
    for (int n = 0; n < 4; ++n) {
      v16bf bf = ldsB(&Ws[n * 16][0], 40);
      acc[n] = wmma_bf16(af, bf, acc[n]);
    }
    __syncthreads();
  }

  const int l = tid & 31, half = l >> 4, nn = l & 15;
#pragma unroll
  for (int j = 0; j < 8; ++j) {
    const int m = m0 + wv * 16 + (half ? j + 8 : j);
#pragma unroll
    for (int n = 0; n < 4; ++n) {
      const int col = n0 + n * 16 + nn;
      Cout[(size_t)m * D_MODEL + col] = acc[n][j] + bias[col];
    }
  }
}

// =====================================================================
extern "C" void kernel_launch(void* const* d_in, const int* in_sizes, int n_in,
                              void* d_out, int out_size, void* d_ws, size_t ws_size,
                              hipStream_t stream) {
  const float* q    = (const float*)d_in[0];
  const float* k    = (const float*)d_in[1];
  const float* v    = (const float*)d_in[2];
  const float* cosp = (const float*)d_in[3];
  const float* sinp = (const float*)d_in[4];
  const unsigned char* mask = (const unsigned char*)d_in[5];
  const float* Wq = (const float*)d_in[6];  const float* bq = (const float*)d_in[7];
  const float* Wk = (const float*)d_in[8];  const float* bk = (const float*)d_in[9];
  const float* Wv = (const float*)d_in[10]; const float* bv = (const float*)d_in[11];
  const float* Wo = (const float*)d_in[12]; const float* bo = (const float*)d_in[13];

  const size_t mat = (size_t)MTOT * D_MODEL;         // 8.4M bf16 elems = 16 MB each
  unsigned short* Qws = (unsigned short*)d_ws;
  unsigned short* Kws = Qws + mat;
  unsigned short* Vws = Kws + mat;
  unsigned short* Ows = Vws + mat;

  dim3 blk(256);
  dim3 g1(MTOT / 128, D_MODEL / 64);
  qkv_rope_kernel<<<g1, blk, 0, stream>>>(q, Wq, bq, cosp, sinp, Qws, 1);
  qkv_rope_kernel<<<g1, blk, 0, stream>>>(k, Wk, bk, cosp, sinp, Kws, 1);
  qkv_rope_kernel<<<g1, blk, 0, stream>>>(v, Wv, bv, cosp, sinp, Vws, 0);

  dim3 g2(TSEQ / 128, NHEAD, BATCH);
  flash_attn_kernel<<<g2, blk, 0, stream>>>(Qws, Kws, Vws, mask, Ows);

  dim3 g3(MTOT / 128, D_MODEL / 64);
  oproj_kernel<<<g3, blk, 0, stream>>>(Ows, Wo, bo, (float*)d_out);
}
